// ResonanceCore_30408368455967
// MI455X (gfx1250) — matrix operand order repaired and checked by reading, not compiled, not run on hardware
//
#include <hip/hip_runtime.h>
#include <hip/hip_bf16.h>
#include <math.h>

// ---------------------------------------------------------------------------
// ResonanceCore for MI455X (gfx1250, wave32, WMMA).
//
//  GEMM1 (B x 256) @ (256 x 96)  -> v_wmma_f32_16x16x32_f16 (f32 accum)
//  LN / GELU(erf) / GEMM2 / phase / 3x3 softmax attention -> fp32 VALU
//
// Layout assumptions (CDNA5 ISA 7.12.2):
//   A (16x32 f16): lane L (M=L%16); lanes<16 hold K {0..7, 16..23},
//                  lanes>=16 hold K {8..15, 24..31}; element e=2v+p.
//   B (32x16 f16): lane L (N=L%16); lanes<16 hold K {0..7, 16..23}, lanes>=16 K {8..15, 24..31},
//                  element e=2v+p, same K order as A.
//   C/D (16x16 f32): VGPR r -> row r (lanes<16) / row r+8 (lanes>=16), col=L%16.
// ---------------------------------------------------------------------------

typedef __attribute__((ext_vector_type(16))) _Float16     v16h;
typedef __attribute__((ext_vector_type(8)))  float        v8f;
typedef __attribute__((ext_vector_type(8)))  unsigned int v8u;

#define DIN          256
#define COLS         96        // H * 2R
#define ROWS_PER_WG  128
#define HSTRIDE      97        // odd stride -> conflict-free column reads
#define NFRAG        48        // 6 column tiles * 8 k-steps
#define FRAG_STRIDE  272       // 32 lanes * 8 dwords + swizzle slack

__device__ __forceinline__ int bfragOff(int frag, int lane) {
    // +4-dword shift per group of 8 lanes: every bank hit exactly 2x per b128.
    return frag * FRAG_STRIDE + lane * 8 + ((lane >> 3) << 2);
}

extern "C" __global__ __launch_bounds__(256)
void resonance_core_kernel(const float* __restrict__ x,
                           const float* __restrict__ W1,
                           const float* __restrict__ b1,
                           const float* __restrict__ lng,
                           const float* __restrict__ lnb,
                           const float* __restrict__ W2,
                           const float* __restrict__ b2,
                           const float* __restrict__ phase,
                           const float* __restrict__ amp,
                           float* __restrict__ outF,
                           float* __restrict__ outS,
                           int nTiles)
{
    __shared__ unsigned int sB[NFRAG * FRAG_STRIDE];      // W1 as f16x2 B-fragments
    __shared__ float sH[ROWS_PER_WG * HSTRIDE];           // GEMM1 output, row major
    __shared__ float sW2[3 * 32 * 16];
    __shared__ float sB1[96], sG[96], sBe[96];
    __shared__ float sB2[48], sCos[48], sSin[48], sAmp[16];

    const int tid  = threadIdx.x;
    const int lane = tid & 31;
    const int wave = tid >> 5;
    const int l16  = lane & 15;
    const int hi   = lane >> 4;

    // ---------------- one-time setup: stage W1 (f16 fragments) + params -----
    for (int idx = tid; idx < NFRAG * 32 * 8; idx += 256) {
        const int d    = idx & 7;
        const int ln   = (idx >> 3) & 31;
        const int frag = idx >> 8;            // frag = t*8 + kk
        const int kk   = frag & 7;
        const int t    = frag >> 3;
        const int n    = t * 16 + (ln & 15);  // output column 0..95
        const int h    = n >> 5;
        const int o    = n & 31;
        const int k0   = kk * 32 + (d >> 2) * 16 + (ln >> 4) * 8 + 2 * (d & 3);
        const float w0 = W1[h * (DIN * 32) + (k0    ) * 32 + o];
        const float w1 = W1[h * (DIN * 32) + (k0 + 1) * 32 + o];
        const unsigned short u0 = __builtin_bit_cast(unsigned short, (_Float16)w0);
        const unsigned short u1 = __builtin_bit_cast(unsigned short, (_Float16)w1);
        sB[bfragOff(frag, ln) + d] = (unsigned int)u0 | ((unsigned int)u1 << 16);
    }
    for (int i = tid; i < 96; i += 256) { sB1[i] = b1[i]; sG[i] = lng[i]; sBe[i] = lnb[i]; }
    for (int i = tid; i < 48; i += 256) {
        sB2[i] = b2[i];
        const float p = phase[i];
        sCos[i] = cosf(p); sSin[i] = sinf(p);
    }
    for (int i = tid; i < 3 * 32 * 16; i += 256) sW2[i] = W2[i];
    if (tid < 16) sAmp[tid] = amp[tid];
    __syncthreads();

    for (int tile = blockIdx.x; tile < nTiles; tile += gridDim.x) {
        const int rowBase = tile * ROWS_PER_WG;

        // ---------------- phase 1: GEMM1 via v_wmma_f32_16x16x32_f16 --------
        {
            const float* xr = x + (size_t)(rowBase + wave * 16 + l16) * DIN + hi * 8;
            v8f acc[6];
            #pragma unroll
            for (int t = 0; t < 6; ++t) acc[t] = (v8f){0.f,0.f,0.f,0.f,0.f,0.f,0.f,0.f};

            #pragma unroll
            for (int kk = 0; kk < 8; ++kk) {
                // A fragment: two contiguous 8-float runs of this row.
                const float4* p = (const float4*)(xr + kk * 32);
                const float4 f0 = p[0], f1 = p[1], f2 = p[4], f3 = p[5];
                v16h a;
                a[0]=(_Float16)f0.x;  a[1]=(_Float16)f0.y;  a[2]=(_Float16)f0.z;  a[3]=(_Float16)f0.w;
                a[4]=(_Float16)f1.x;  a[5]=(_Float16)f1.y;  a[6]=(_Float16)f1.z;  a[7]=(_Float16)f1.w;
                a[8]=(_Float16)f2.x;  a[9]=(_Float16)f2.y;  a[10]=(_Float16)f2.z; a[11]=(_Float16)f2.w;
                a[12]=(_Float16)f3.x; a[13]=(_Float16)f3.y; a[14]=(_Float16)f3.z; a[15]=(_Float16)f3.w;

                #pragma unroll
                for (int t = 0; t < 6; ++t) {
                    const int off = bfragOff(t * 8 + kk, lane);
                    const uint4 lo = *(const uint4*)&sB[off];
                    const uint4 hi4 = *(const uint4*)&sB[off + 4];
                    v8u bu;
                    bu[0]=lo.x; bu[1]=lo.y; bu[2]=lo.z; bu[3]=lo.w;
                    bu[4]=hi4.x; bu[5]=hi4.y; bu[6]=hi4.z; bu[7]=hi4.w;
                    const v16h bfr = __builtin_bit_cast(v16h, bu);
                    acc[t] = __builtin_amdgcn_wmma_f32_16x16x32_f16(
                        false, a, false, bfr, (short)0, acc[t], false, false);
                }
            }
            // Scatter C fragments to row-major LDS.
            #pragma unroll
            for (int t = 0; t < 6; ++t) {
                const int col = t * 16 + l16;
                const int r0  = wave * 16 + hi * 8;
                #pragma unroll
                for (int r = 0; r < 8; ++r)
                    sH[(r0 + r) * HSTRIDE + col] = acc[t][r];
            }
        }
        __syncthreads();

        // ---------------- phase 2: per-row LN/GELU/GEMM2/attention ----------
        if (tid < ROWS_PER_WG) {
            const int row = rowBase + tid;
            const float* hrow = &sH[tid * HSTRIDE];
            float hh[3][16];

            #pragma unroll
            for (int h = 0; h < 3; ++h) {
                float s = 0.f, s2 = 0.f;
                for (int o = 0; o < 32; ++o) {
                    const float v = hrow[h * 32 + o] + sB1[h * 32 + o];
                    s += v; s2 += v * v;
                }
                const float mu  = s * 0.03125f;
                const float var = s2 * 0.03125f - mu * mu;   // biased, as torch LN
                const float rs  = 1.0f / sqrtf(var + 1e-5f);

                float acc[16];
                #pragma unroll
                for (int r = 0; r < 16; ++r) acc[r] = 0.f;
                for (int o = 0; o < 32; ++o) {
                    const int c = h * 32 + o;
                    float v = (hrow[c] + sB1[c] - mu) * rs;
                    v = v * sG[c] + sBe[c];
                    const float g = 0.5f * v * (1.0f + erff(v * 0.70710678118654752f));
                    const float* w2 = &sW2[c * 16];
                    #pragma unroll
                    for (int r = 0; r < 16; ++r) acc[r] += g * w2[r];
                }
                #pragma unroll
                for (int r = 0; r < 16; ++r) hh[h][r] = acc[r] + sB2[h * 16 + r];
            }

            // phase modulation: h*cos + roll(h,1)*sin  (rolled[r] = h[r-1])
            #pragma unroll
            for (int h = 0; h < 3; ++h) {
                float tmp[16];
                #pragma unroll
                for (int r = 0; r < 16; ++r)
                    tmp[r] = hh[h][r] * sCos[h * 16 + r]
                           + hh[h][(r + 15) & 15] * sSin[h * 16 + r];
                #pragma unroll
                for (int r = 0; r < 16; ++r) hh[h][r] = tmp[r];
            }

            // 3x3 attention over harmonics
            float rm[3][3];
            #pragma unroll
            for (int a = 0; a < 3; ++a)
                #pragma unroll
                for (int k = 0; k < 3; ++k) {
                    float s = 0.f;
                    #pragma unroll
                    for (int d = 0; d < 16; ++d) s += hh[a][d] * hh[k][d];
                    rm[a][k] = s;
                }
            float sc[3][3];
            #pragma unroll
            for (int a = 0; a < 3; ++a) {
                const float m  = fmaxf(rm[a][0], fmaxf(rm[a][1], rm[a][2]));
                const float e0 = expf(rm[a][0] - m);
                const float e1 = expf(rm[a][1] - m);
                const float e2 = expf(rm[a][2] - m);
                const float inv = 1.0f / (e0 + e1 + e2);
                sc[a][0] = e0 * inv; sc[a][1] = e1 * inv; sc[a][2] = e2 * inv;
            }

            float* os = outS + (size_t)row * 9;
            #pragma unroll
            for (int a = 0; a < 3; ++a) {
                os[a * 3 + 0] = sc[a][0];
                os[a * 3 + 1] = sc[a][1];
                os[a * 3 + 2] = sc[a][2];
            }
            float* of = outF + (size_t)row * 16;
            #pragma unroll
            for (int d = 0; d < 16; ++d) {
                float f = 0.f;
                #pragma unroll
                for (int a = 0; a < 3; ++a) {
                    const float th = sc[a][0] * hh[0][d]
                                   + sc[a][1] * hh[1][d]
                                   + sc[a][2] * hh[2][d];
                    f += hh[a][d] * th;
                }
                of[d] = f * sAmp[d];
            }
        }
        __syncthreads();   // protect sH before next tile overwrites it
    }
}

extern "C" void kernel_launch(void* const* d_in, const int* in_sizes, int n_in,
                              void* d_out, int out_size, void* d_ws, size_t ws_size,
                              hipStream_t stream) {
    const float* x   = (const float*)d_in[0];
    const float* W1  = (const float*)d_in[1];
    const float* b1  = (const float*)d_in[2];
    const float* lng = (const float*)d_in[3];
    const float* lnb = (const float*)d_in[4];
    const float* W2  = (const float*)d_in[5];
    const float* b2  = (const float*)d_in[6];
    const float* ph  = (const float*)d_in[7];
    const float* amp = (const float*)d_in[8];

    const int B      = in_sizes[0] / DIN;       // 262144
    const int nTiles = B / ROWS_PER_WG;         // 2048 (B is a multiple of 128)
    float* outF = (float*)d_out;                // field: B*16 floats
    float* outS = outF + (size_t)B * 16;        // score: B*9 floats

    const int blocks = (nTiles < 1024) ? nTiles : 1024;
    resonance_core_kernel<<<dim3(blocks), dim3(256), 0, stream>>>(
        x, W1, b1, lng, lnb, W2, b2, ph, amp, outF, outS, nTiles);
}
